// Z9QATAttention_47132971106622
// MI455X (gfx1250) — compile-verified, matrix operand order checked by reading, not run on hardware
//
#include <hip/hip_runtime.h>
#include <hip/hip_bf16.h>
#include <math.h>

typedef __attribute__((ext_vector_type(16))) __bf16 v16bf;
typedef __attribute__((ext_vector_type(8)))  __bf16 v8bf;
typedef __attribute__((ext_vector_type(8)))  float  v8f;
typedef __attribute__((ext_vector_type(4)))  unsigned su4;
typedef __attribute__((ext_vector_type(8)))  unsigned su8;

static constexpr int Bsz = 2, T = 2048, Cdim = 1024, H = 16, D = 64;

__device__ __forceinline__ v8f wmma_bf16(v16bf a, v16bf b, v8f c) {
  return __builtin_amdgcn_wmma_f32_16x16x32_bf16(false, a, false, b, (short)0, c, false, false);
}

// Async global->LDS 16-byte copy (ASYNCcnt path). LDS address is the low 32
// bits of the generic pointer (ISA 10.2).
__device__ __forceinline__ void async_ld16(const void* lds, const void* g) {
  asm volatile("global_load_async_to_lds_b128 %0, %1, off"
               :: "v"((unsigned)(size_t)lds),
                  "v"((unsigned long long)(size_t)g)
               : "memory");
}
__device__ __forceinline__ void wait_async0() {
  asm volatile("s_wait_asynccnt 0x0" ::: "memory");
}

// Tensor Data Mover: DMA one 64x64 bf16 tile (row stride 64 elems in memory)
// into LDS with a 16B pad after every 128B row => LDS row stride 72 elems.
// D# per ISA 8.3/8.4: group0 {count=1, lds_addr, gaddr lo, gaddr hi | type=2};
// group1 {data_size=2B, pad_enable, pad_interval=4 (32 DW), pad_amount=3 (4 DW),
//         tensor_dim0=64, tensor_dim1=64, tile_dim0=64, tile_dim1=64,
//         tensor_dim0_stride=64}. Groups 2/3 zero (tile_dim2=0 -> unused).
__device__ __forceinline__ void tdm_load_tile64x64(const void* lds, const void* gsrc) {
  const unsigned long long ga = (unsigned long long)(size_t)gsrc;
  su4 g0; su8 g1; su4 gz;
  g0[0] = 1u;                                   // count=1 (valid user D#)
  g0[1] = (unsigned)(size_t)lds;                // lds_addr (bytes)
  g0[2] = (unsigned)ga;                         // global_addr[31:0]
  g0[3] = (unsigned)(ga >> 32) | (2u << 30);    // global_addr[56:32] | type=2
  g1[0] = (1u << 16) | (1u << 20) | (4u << 22) | (3u << 25);
  g1[1] = 64u << 16;                            // tensor_dim0 = 64
  g1[2] = 64u << 16;                            // tensor_dim1 = 64
  g1[3] = 64u << 16;                            // tile_dim0 = 64
  g1[4] = 64u;                                  // tile_dim1 = 64 (tile_dim2=0)
  g1[5] = 64u;                                  // tensor_dim0_stride = 64
  g1[6] = 0u;
  g1[7] = 0u;
  gz[0] = 0u; gz[1] = 0u; gz[2] = 0u; gz[3] = 0u;
  asm volatile("tensor_load_to_lds %0, %1, %2, %3"
               :: "s"(g0), "s"(g1), "s"(gz), "s"(gz)
               : "memory");
}
__device__ __forceinline__ void wait_tensor0() {
#if __has_builtin(__builtin_amdgcn_s_wait_tensorcnt)
  __builtin_amdgcn_s_wait_tensorcnt(0);
#else
  asm volatile("s_wait_tensorcnt 0x0" ::: "memory");
#endif
}

// Load a 16x32 bf16 fragment (A-layout; also B-layout when the LDS tile is
// stored [n][k]). Per ISA 7.12.2: lane l holds row (l&15); its 16 values are
// K = {g*16 + (l>>4)*8 + i : g in 0..1, i in 0..7}.
template<int STRIDE>
__device__ __forceinline__ v16bf load_frag(const __bf16* base, int rowbase, int kbase, int lane) {
  const int row  = rowbase + (lane & 15);
  const int half = lane >> 4;
  const __bf16* p = base + row * STRIDE + kbase + half * 8;
  v8bf lo = *(const v8bf*)p;
  v8bf hi = *(const v8bf*)(p + 16);
  return __builtin_shufflevector(lo, hi, 0,1,2,3,4,5,6,7,8,9,10,11,12,13,14,15);
}

// ---------------------------------------------------------------------------
// Prep 1: elementwise fp32 -> bf16 (x). n must be a multiple of 2048.
// ---------------------------------------------------------------------------
__global__ __launch_bounds__(256) void convert_bf16(const float* __restrict__ in,
                                                    __bf16* __restrict__ out, int n) {
  const int i = (blockIdx.x * 256 + threadIdx.x) * 8;
  if (i >= n) return;
  const float4* p = (const float4*)(in + i);
  float4 a = p[0], b = p[1];
  v8bf o;
  o[0] = (__bf16)a.x; o[1] = (__bf16)a.y; o[2] = (__bf16)a.z; o[3] = (__bf16)a.w;
  o[4] = (__bf16)b.x; o[5] = (__bf16)b.y; o[6] = (__bf16)b.z; o[7] = (__bf16)b.w;
  *(v8bf*)(out + i) = o;
}

// ---------------------------------------------------------------------------
// Prep 2: transpose + convert weights: in[K][N] fp32 -> out[N][K] bf16.
// 32x32 LDS tile, 256 threads. Grid: (N/32, K/32).
// ---------------------------------------------------------------------------
__global__ __launch_bounds__(256) void transpose_to_bf16(const float* __restrict__ in,
                                                         __bf16* __restrict__ out,
                                                         int K, int N) {
  __shared__ float tile[32][33];
  const int tx = threadIdx.x & 31, ty = threadIdx.x >> 5;   // ty: 0..7
  const int n0 = blockIdx.x * 32, k0 = blockIdx.y * 32;
#pragma unroll
  for (int i = 0; i < 4; ++i)
    tile[ty + i * 8][tx] = in[(size_t)(k0 + ty + i * 8) * N + n0 + tx];
  __syncthreads();
#pragma unroll
  for (int i = 0; i < 4; ++i)
    out[(size_t)(n0 + ty + i * 8) * K + k0 + tx] = (__bf16)tile[tx][ty + i * 8];
}

// ---------------------------------------------------------------------------
// Kernel 1: qkv = xb @ wqkvT^T  (both bf16, f32 accum). M=4096, N=3072, K=1024.
// 128x128 tile, K-step 32, async-staged, double-buffered LDS.
// Writes Q/K/V as bf16 in [B][H][T][D].
// ---------------------------------------------------------------------------
__global__ __launch_bounds__(256) void qkv_gemm(const __bf16* __restrict__ xb,
                                                const __bf16* __restrict__ wT,
                                                __bf16* __restrict__ qo,
                                                __bf16* __restrict__ ko,
                                                __bf16* __restrict__ vo) {
  __shared__ __bf16 As[2][128 * 40];   // [m][k], 80B row stride (16B aligned)
  __shared__ __bf16 Bs[2][128 * 40];   // [n][k]

  const int tid  = threadIdx.x;
  const int lane = tid & 31;
  const int wid  = tid >> 5;
  const int wm   = wid >> 2;        // 0..1 -> 64 rows
  const int wn   = wid & 3;         // 0..3 -> 32 cols
  const int m0   = blockIdx.y * 128;
  const int n0   = blockIdx.x * 128;

  v8f acc[4][2];
#pragma unroll
  for (int i = 0; i < 4; ++i)
#pragma unroll
    for (int j = 0; j < 2; ++j)
#pragma unroll
      for (int r = 0; r < 8; ++r) acc[i][j][r] = 0.f;

  const int srow = tid >> 1;          // 0..127
  const int sc   = (tid & 1) * 16;    // 0 / 16

  auto stage = [&](int buf, int k0) {
    const __bf16* ga = xb + (size_t)(m0 + srow) * Cdim + k0 + sc;
    const __bf16* gb = wT + (size_t)(n0 + srow) * Cdim + k0 + sc;
    async_ld16(&As[buf][srow * 40 + sc],     ga);
    async_ld16(&As[buf][srow * 40 + sc + 8], ga + 8);
    async_ld16(&Bs[buf][srow * 40 + sc],     gb);
    async_ld16(&Bs[buf][srow * 40 + sc + 8], gb + 8);
  };

  stage(0, 0);
  wait_async0();
  __syncthreads();

  int cur = 0;
  for (int k0 = 0; k0 < Cdim; k0 += 32) {
    const int nxt = cur ^ 1;
    if (k0 + 32 < Cdim) stage(nxt, k0 + 32);

    v16bf af[4];
#pragma unroll
    for (int i = 0; i < 4; ++i) af[i] = load_frag<40>(As[cur], wm * 64 + i * 16, 0, lane);
    v16bf bfr[2];
#pragma unroll
    for (int j = 0; j < 2; ++j) bfr[j] = load_frag<40>(Bs[cur], wn * 32 + j * 16, 0, lane);
#pragma unroll
    for (int i = 0; i < 4; ++i)
#pragma unroll
      for (int j = 0; j < 2; ++j) acc[i][j] = wmma_bf16(af[i], bfr[j], acc[i][j]);

    wait_async0();       // next buffer landed
    __syncthreads();     // everyone done reading cur
    cur = nxt;
  }

  // epilogue: scatter to Q/K/V [B][H][T][D] as bf16
  const int hlf = lane >> 4, ln = lane & 15;
#pragma unroll
  for (int i = 0; i < 4; ++i)
#pragma unroll
    for (int j = 0; j < 2; ++j)
#pragma unroll
      for (int r = 0; r < 8; ++r) {
        const int m = m0 + wm * 64 + i * 16 + r + 8 * hlf;   // b*T + t
        const int n = n0 + wn * 32 + j * 16 + ln;            // which*C + h*D + d
        const int which = n >> 10;
        const int h = (n >> 6) & 15;
        const int d = n & 63;
        const int b = m >> 11;
        const int t = m & 2047;
        __bf16* dst = (which == 0) ? qo : ((which == 1) ? ko : vo);
        dst[((size_t)(b * H + h) * T + t) * D + d] = (__bf16)acc[i][j][r];
      }
}

// ---------------------------------------------------------------------------
// Kernel 2: flash attention per (64 q-rows, head). Causal, online softmax.
// Q/K tiles staged by the Tensor Data Mover (one descriptor per tile, issued
// by wave 0; pad feature reproduces the 72-element LDS row stride). V is
// transposed into LDS by all waves while the DMA runs.
// ---------------------------------------------------------------------------
__global__ __launch_bounds__(256) void attention(const __bf16* __restrict__ qg,
                                                 const __bf16* __restrict__ kg,
                                                 const __bf16* __restrict__ vg,
                                                 const float* __restrict__ cw,
                                                 __bf16* __restrict__ att) {
  __shared__ __bf16 Qs[64 * 72];   // [q][d]
  __shared__ __bf16 Ks[64 * 72];   // [key][d]  == B-fragment layout for Q*K^T
  __shared__ __bf16 Vt[64 * 72];   // [d][key]  == B-fragment layout for P*V
  __shared__ __bf16 Pb[64 * 72];   // [q][key]  bf16 probabilities
  __shared__ float  Sb[64 * 65];   // [q][key]  f32 scores
  __shared__ float  mrow[64], lrow[64], frow[64];

  const int tid = threadIdx.x, lane = tid & 31, wid = tid >> 5;
  const int qi = blockIdx.x;            // q-tile (0..31)
  const int q0 = qi * 64;
  const int bh = blockIdx.y;            // b*H + h
  const int b = bh >> 4, h = bh & 15;
  const size_t base = (size_t)bh * T * D;

  const float scale = 0.125f;           // D^-0.5, D=64
  const float bias  = fmodf(cw[0] + cw[1] + cw[2], 9.0f) * 0.001f;

  const int row = tid >> 2, c0 = (tid & 3) * 16;

  // Q tile via TDM (completion published by wave 0's tensorcnt wait below)
  if (wid == 0) tdm_load_tile64x64(&Qs[0], qg + base + (size_t)q0 * D);
  if (tid < 64) { mrow[tid] = -3.0e38f; lrow[tid] = 0.f; }

  v8f oacc[2];
#pragma unroll
  for (int j = 0; j < 2; ++j)
#pragma unroll
    for (int r = 0; r < 8; ++r) oacc[j][r] = 0.f;

  const int sm  = wid >> 1;             // q sub-tile (0..3)
  const int sn2 = (wid & 1) * 2;        // first of two key/d sub-tiles
  const int hlf = lane >> 4, ln = lane & 15;

  for (int j = 0; j <= qi; ++j) {
    __syncthreads();                    // prior iter done with Ks/Vt/Pb
    // K tile DMA; V tile transposed by all waves while the DMA runs
    if (wid == 0) tdm_load_tile64x64(&Ks[0], kg + base + (size_t)(j * 64) * D);
    {
      const uint4* sv = (const uint4*)(vg + base + (size_t)(j * 64 + row) * D + c0);
      __bf16 tmp[16];
      *(uint4*)&tmp[0] = sv[0];
      *(uint4*)&tmp[8] = sv[1];
#pragma unroll
      for (int i = 0; i < 16; ++i) Vt[(c0 + i) * 72 + row] = tmp[i];
    }
    if (wid == 0) wait_tensor0();       // Q (first iter) + K complete
    __syncthreads();                    // publishes DMA'd tiles to all waves

    // S = (Q K^T) * scale + bias, causal-masked
    v8f sacc[2];
#pragma unroll
    for (int jj = 0; jj < 2; ++jj)
#pragma unroll
      for (int r = 0; r < 8; ++r) sacc[jj][r] = 0.f;
#pragma unroll
    for (int kk = 0; kk < 2; ++kk) {
      v16bf aQ = load_frag<72>(Qs, sm * 16, kk * 32, lane);
#pragma unroll
      for (int jj = 0; jj < 2; ++jj) {
        v16bf bK = load_frag<72>(Ks, (sn2 + jj) * 16, kk * 32, lane);
        sacc[jj] = wmma_bf16(aQ, bK, sacc[jj]);
      }
    }
#pragma unroll
    for (int jj = 0; jj < 2; ++jj)
#pragma unroll
      for (int r = 0; r < 8; ++r) {
        const int qr = sm * 16 + r + 8 * hlf;
        const int kc = (sn2 + jj) * 16 + ln;
        float val = sacc[jj][r] * scale + bias;
        if (j * 64 + kc > q0 + qr) val = -1.0e9f;
        Sb[qr * 65 + kc] = val;
      }
    __syncthreads();

    // online softmax: one thread per q-row
    if (tid < 64) {
      const float mold = mrow[tid];
      float mx = mold;
      for (int c = 0; c < 64; ++c) mx = fmaxf(mx, Sb[tid * 65 + c]);
      const float f = __expf(mold - mx);
      float sum = 0.f;
      for (int c = 0; c < 64; ++c) {
        const float p = __expf(Sb[tid * 65 + c] - mx);
        Pb[tid * 72 + c] = (__bf16)p;
        sum += p;
      }
      mrow[tid] = mx;
      lrow[tid] = lrow[tid] * f + sum;
      frow[tid] = f;
    }
    __syncthreads();

    // rescale O and accumulate O += P @ V
#pragma unroll
    for (int jj = 0; jj < 2; ++jj)
#pragma unroll
      for (int r = 0; r < 8; ++r) {
        const int qr = sm * 16 + r + 8 * hlf;
        oacc[jj][r] *= frow[qr];
      }
#pragma unroll
    for (int kk = 0; kk < 2; ++kk) {
      v16bf aP = load_frag<72>(Pb, sm * 16, kk * 32, lane);
#pragma unroll
      for (int jj = 0; jj < 2; ++jj) {
        v16bf bV = load_frag<72>(Vt, (sn2 + jj) * 16, kk * 32, lane);
        oacc[jj] = wmma_bf16(aP, bV, oacc[jj]);
      }
    }
  }

  // normalize and write attention output as bf16 [B][T][C] (C = h*64 + d)
#pragma unroll
  for (int jj = 0; jj < 2; ++jj)
#pragma unroll
    for (int r = 0; r < 8; ++r) {
      const int qr = sm * 16 + r + 8 * hlf;
      const int d  = (sn2 + jj) * 16 + ln;
      const float val = oacc[jj][r] / lrow[qr];
      att[((size_t)b * T + q0 + qr) * Cdim + h * D + d] = (__bf16)val;
    }
}

// ---------------------------------------------------------------------------
// Kernel 3: out = att(bf16) @ woutT^T (bf16), fp32 result. M=4096, N=K=1024.
// Async-staged, double-buffered, same structure as kernel 1.
// ---------------------------------------------------------------------------
__global__ __launch_bounds__(256) void out_gemm(const __bf16* __restrict__ att,
                                                const __bf16* __restrict__ wT,
                                                float* __restrict__ out) {
  __shared__ __bf16 As[2][128 * 40];
  __shared__ __bf16 Bs[2][128 * 40];

  const int tid  = threadIdx.x;
  const int lane = tid & 31;
  const int wid  = tid >> 5;
  const int wm   = wid >> 2;
  const int wn   = wid & 3;
  const int m0   = blockIdx.y * 128;
  const int n0   = blockIdx.x * 128;

  v8f acc[4][2];
#pragma unroll
  for (int i = 0; i < 4; ++i)
#pragma unroll
    for (int j = 0; j < 2; ++j)
#pragma unroll
      for (int r = 0; r < 8; ++r) acc[i][j][r] = 0.f;

  const int srow = tid >> 1;
  const int sc   = (tid & 1) * 16;

  auto stage = [&](int buf, int k0) {
    const __bf16* ga = att + (size_t)(m0 + srow) * Cdim + k0 + sc;
    const __bf16* gb = wT  + (size_t)(n0 + srow) * Cdim + k0 + sc;
    async_ld16(&As[buf][srow * 40 + sc],     ga);
    async_ld16(&As[buf][srow * 40 + sc + 8], ga + 8);
    async_ld16(&Bs[buf][srow * 40 + sc],     gb);
    async_ld16(&Bs[buf][srow * 40 + sc + 8], gb + 8);
  };

  stage(0, 0);
  wait_async0();
  __syncthreads();

  int cur = 0;
  for (int k0 = 0; k0 < Cdim; k0 += 32) {
    const int nxt = cur ^ 1;
    if (k0 + 32 < Cdim) stage(nxt, k0 + 32);

    v16bf af[4];
#pragma unroll
    for (int i = 0; i < 4; ++i) af[i] = load_frag<40>(As[cur], wm * 64 + i * 16, 0, lane);
    v16bf bfr[2];
#pragma unroll
    for (int j = 0; j < 2; ++j) bfr[j] = load_frag<40>(Bs[cur], wn * 32 + j * 16, 0, lane);
#pragma unroll
    for (int i = 0; i < 4; ++i)
#pragma unroll
      for (int j = 0; j < 2; ++j) acc[i][j] = wmma_bf16(af[i], bfr[j], acc[i][j]);

    wait_async0();
    __syncthreads();
    cur = nxt;
  }

  const int hlf = lane >> 4, ln = lane & 15;
#pragma unroll
  for (int i = 0; i < 4; ++i)
#pragma unroll
    for (int j = 0; j < 2; ++j)
#pragma unroll
      for (int r = 0; r < 8; ++r) {
        const int m = m0 + wm * 64 + i * 16 + r + 8 * hlf;
        const int n = n0 + wn * 32 + j * 16 + ln;
        out[(size_t)m * Cdim + n] = acc[i][j][r];
      }
}

// ---------------------------------------------------------------------------
extern "C" void kernel_launch(void* const* d_in, const int* in_sizes, int n_in,
                              void* d_out, int out_size, void* d_ws, size_t ws_size,
                              hipStream_t stream) {
  (void)in_sizes; (void)n_in; (void)out_size; (void)ws_size;
  const float* x    = (const float*)d_in[0];
  // d_in[1] = tril mask (causality computed analytically; identical semantics)
  const float* wqkv = (const float*)d_in[2];
  const float* wout = (const float*)d_in[3];
  const float* cw   = (const float*)d_in[4];
  float* out = (float*)d_out;

  const size_t nX   = (size_t)Bsz * T * Cdim;    // 4,194,304
  const size_t nWq  = (size_t)Cdim * 3 * Cdim;   // 3,145,728
  const size_t nWo  = (size_t)Cdim * Cdim;       // 1,048,576
  const size_t nQKV = (size_t)Bsz * H * T * D;   // 4,194,304

  __bf16* xb    = (__bf16*)d_ws;         // dead after qkv_gemm
  __bf16* wqkvT = xb + nX;
  __bf16* woutT = wqkvT + nWq;
  __bf16* q     = woutT + nWo;
  __bf16* k     = q + nQKV;
  __bf16* v     = k + nQKV;
  __bf16* att   = xb;                    // alias: reuses xb region

  convert_bf16     <<<nX / 2048, 256, 0, stream>>>(x, xb, (int)nX);
  transpose_to_bf16<<<dim3(3 * Cdim / 32, Cdim / 32), 256, 0, stream>>>(wqkv, wqkvT, Cdim, 3 * Cdim);
  transpose_to_bf16<<<dim3(Cdim / 32, Cdim / 32),     256, 0, stream>>>(wout, woutT, Cdim, Cdim);

  qkv_gemm <<<dim3(3 * Cdim / 128, Bsz * T / 128), 256, 0, stream>>>(xb, wqkvT, q, k, v);
  attention<<<dim3(T / 64, Bsz * H),               256, 0, stream>>>(q, k, v, cw, att);
  out_gemm <<<dim3(Cdim / 128, Bsz * T / 128),     256, 0, stream>>>(att, woutT, out);
}
